// BF16AccumConv2d_70171175682828
// MI455X (gfx1250) — matrix operand order repaired and checked by reading, not compiled, not run on hardware
//
#include <hip/hip_runtime.h>

// ---- problem constants (from reference) ------------------------------------
#define B_     8
#define CIN    64
#define HH     128
#define WW     128
#define COUT   64
#define HOUT   126
#define WOUT   126
#define HWIN   (HH * WW)                  // 16384
#define CHW    (CIN * HWIN)
#define NXELEM (B_ * CHW)                 // 8388608 x elements
#define XBF_BYTES ((size_t)NXELEM * 2)    // 16 MiB NHWC bf16 copy of x in ws
#define OWT    4                          // 32-wide ow tiles per row
#define TILES  (B_ * HOUT * OWT)          // 4032 wave tiles == 1008 blocks * 4 waves
#define WT_ELEMS (9 * COUT * CIN)         // 36864 bf16 = 72 KiB (all weight slices)

typedef __attribute__((ext_vector_type(16))) __bf16         v16bf;
typedef __attribute__((ext_vector_type(16))) unsigned short v16u;
typedef __attribute__((ext_vector_type(8)))  unsigned short v8u;
typedef __attribute__((ext_vector_type(8)))  float          v8f;
typedef __attribute__((ext_vector_type(4)))  unsigned       v4u;

__device__ __forceinline__ unsigned short f32_bf16(float f) {
  unsigned u = __builtin_bit_cast(unsigned, f);
  u += 0x7FFFu + ((u >> 16) & 1u);              // round-to-nearest-even
  return (unsigned short)(u >> 16);
}
__device__ __forceinline__ unsigned pack_bf16x2(float lo, float hi) {
  return (unsigned)f32_bf16(lo) | ((unsigned)f32_bf16(hi) << 16);
}

// ---- prep kernel 1: x NCHW f32 -> NHWC bf16 (channel contiguous) ------------
__launch_bounds__(256)
__global__ void x_to_nhwc_bf16(const float* __restrict__ x, unsigned short* __restrict__ xt) {
  const int t   = threadIdx.x;
  const int g   = t & 7;            // channel group (8 channels)
  const int iwl = t >> 3;           // 0..31
  const int bz  = blockIdx.x;       // b * 128 ih * 4 iw-tiles = 4096
  const int iwt = bz & 3;
  const int ihb = bz >> 2;
  const int ih  = ihb & (HH - 1);
  const int bI  = ihb >> 7;
  const int iw  = iwt * 32 + iwl;
  const int c0  = g * 8;

  const float* src = x + ((size_t)(bI * CIN + c0) * HH + ih) * WW + iw;
  float f[8];
#pragma unroll
  for (int j = 0; j < 8; ++j) f[j] = src[(size_t)j * HWIN];   // immediate offsets

  v4u r;
  r[0] = pack_bf16x2(f[0], f[1]);
  r[1] = pack_bf16x2(f[2], f[3]);
  r[2] = pack_bf16x2(f[4], f[5]);
  r[3] = pack_bf16x2(f[6], f[7]);
  *(v4u*)(xt + ((size_t)(bI * HH + ih) * WW + iw) * CIN + c0) = r;   // dense b128 store
}

// ---- prep kernel 2: weight [n][c][kh][kw] -> wt [kh*3+kw][n][c] -------------
__launch_bounds__(64)
__global__ void wt_transpose(const unsigned short* __restrict__ w, unsigned short* __restrict__ wt) {
  const int n = blockIdx.x;     // 64
  const int c = threadIdx.x;    // 64
#pragma unroll
  for (int kk = 0; kk < 9; ++kk)
    wt[((size_t)kk * COUT + n) * CIN + c] = w[((size_t)n * CIN + c) * 9 + kk];
}

// ---- main kernel: implicit-GEMM conv, one wave = 32 ow x 64 cout tile -------
// Weights (72 KiB, transposed) staged into dynamic LDS once; K loop is pure
// loads + WMMA with zero barriers. Register blocking: each B fragment feeds
// two WMMAs (M-tiles), halving LDS traffic and dscnt waits per matrix op.
__launch_bounds__(128)
__global__ void conv3x3_bf16_wmma(const unsigned short* __restrict__ xt,
                                  const unsigned short* __restrict__ wt,
                                  const unsigned short* __restrict__ bias,
                                  unsigned short* __restrict__ out) {
  extern __shared__ __align__(16) unsigned short ldsW[];   // 73728 B at launch

  const int tid  = threadIdx.x;
  const int lane = tid & 31;
  const int wave = tid >> 5;
  const int half = lane >> 4;
  const int l16  = lane & 15;

  // ---- one-shot staging of all 9 weight slices: 36 x 16B per thread ----------
  {
    const v4u* src = (const v4u*)wt;
    v4u*       dst = (v4u*)ldsW;
#pragma unroll
    for (int i = 0; i < 36; ++i) dst[tid + i * 128] = src[tid + i * 128];
  }

  const int t   = blockIdx.x * 4 + wave;     // grid covers TILES exactly (no idle waves)
  const int owt = t & 3;
  const int ohb = t >> 2;                    // bI*126 + oh
  const int oh  = ohb % HOUT;
  const int bI  = ohb / HOUT;
  const int owb = owt * 32;

  // NHWC lane base addresses for the two 16-row M tiles ("+ half*16" bakes in
  // this code's contiguous k order, half h holding channels 16h..16h+15).
  const unsigned short* xa[2];
#pragma unroll
  for (int mt = 0; mt < 2; ++mt) {
    const int ow  = owb + mt * 16 + l16;
    const int owc = (ow < WOUT) ? ow : (WOUT - 1);   // clamp ragged tile: no OOB reads
    xa[mt] = xt + ((size_t)(bI * HH + oh) * WW + owc) * CIN + half * 16;
  }

  v8f acc[2][4] = {};

  __syncthreads();                           // single barrier: weights resident

#pragma unroll
  for (int kk = 0; kk < 9; ++kk) {
    const int kh = kk / 3, kw = kk % 3;      // compile-time under unroll
    const size_t off = (size_t)(kh * WW + kw) * CIN;

    // ---- A fragments: 2 M-tiles x 2 c0-chunks, each = two aligned b128 loads.
    // Channels c0+16h+{0..7} (V0-3) and c0+16h+8+{0..7} (V4-7) for half h are
    // contiguous 8-element runs in NHWC -> immediate-offset global_load_b128.
    v16bf af[2][2];
#pragma unroll
    for (int mt = 0; mt < 2; ++mt) {
      const unsigned short* pa = xa[mt] + off;
#pragma unroll
      for (int c0i = 0; c0i < 2; ++c0i) {
        v8u lo = *(const v8u*)(pa + c0i * 32);
        v8u hi = *(const v8u*)(pa + c0i * 32 + 8);
        v16u au = __builtin_shufflevector(lo, hi, 0, 1, 2, 3, 4, 5, 6, 7,
                                                  8, 9, 10, 11, 12, 13, 14, 15);
        af[mt][c0i] = __builtin_bit_cast(v16bf, au);
      }
    }

    // ---- B fragments from the resident LDS weight bank; each feeds 2 WMMAs.
    const unsigned short* lw = ldsW + kk * (COUT * CIN);
#pragma unroll
    for (int c0i = 0; c0i < 2; ++c0i)
#pragma unroll
      for (int nt = 0; nt < 4; ++nt) {
        const int n = nt * 16 + l16;
        v16u  bu    = *(const v16u*)&lw[n * CIN + c0i * 32 + half * 16];  // 16B-aligned pair
        v16bf bfrag = __builtin_bit_cast(v16bf, bu);
#pragma unroll
        for (int mt = 0; mt < 2; ++mt)
          acc[mt][nt] = __builtin_amdgcn_wmma_f32_16x16x32_bf16(
              false, af[mt][c0i], false, bfrag, (short)0, acc[mt][nt], false, false);
      }
  }

  // Epilogue: bias add, bf16 pack, vectorized stores.
  // D layout: lane holds channel n = nt*16+l16; VGPR r holds ow = mbase + r.
#pragma unroll
  for (int mt = 0; mt < 2; ++mt) {
    const int mbase = owb + mt * 16 + half * 8;
#pragma unroll
    for (int nt = 0; nt < 4; ++nt) {
      const int   n  = nt * 16 + l16;
      const float bf = __builtin_bit_cast(float, (unsigned)bias[n] << 16);
      const v8f   a  = acc[mt][nt];
      const size_t ob = (((size_t)bI * COUT + n) * HOUT + oh) * WOUT;
      if (mbase + 8 <= WOUT) {
        unsigned* po = (unsigned*)(out + ob + mbase);   // dword-aligned
#pragma unroll
        for (int p = 0; p < 4; ++p)
          po[p] = pack_bf16x2(a[2 * p] + bf, a[2 * p + 1] + bf);
      } else {
#pragma unroll
        for (int r = 0; r < 8; ++r)
          if (mbase + r < WOUT) out[ob + mbase + r] = f32_bf16(a[r] + bf);
      }
    }
  }
}

extern "C" void kernel_launch(void* const* d_in, const int* in_sizes, int n_in,
                              void* d_out, int out_size, void* d_ws, size_t ws_size,
                              hipStream_t stream) {
  (void)in_sizes; (void)n_in; (void)out_size; (void)ws_size;
  const float*          x    = (const float*)d_in[0];
  const unsigned short* w    = (const unsigned short*)d_in[1];   // bf16 bits
  const unsigned short* bias = (const unsigned short*)d_in[2];   // bf16 bits
  unsigned short*       out  = (unsigned short*)d_out;           // bf16 bits

  unsigned short* xt = (unsigned short*)d_ws;                       // 16 MiB NHWC bf16
  unsigned short* wt = (unsigned short*)((char*)d_ws + XBF_BYTES);  // 72 KiB

  x_to_nhwc_bf16<<<B_ * HH * 4, 256, 0, stream>>>(x, xt);           // 4096 blocks
  wt_transpose<<<COUT, CIN, 0, stream>>>(w, wt);
  conv3x3_bf16_wmma<<<TILES / 4, 128, (unsigned)(WT_ELEMS * sizeof(unsigned short)),
                      stream>>>(xt, wt, bias, out);                 // 1008 blocks, 72 KiB LDS
}
